// CausalSelfAttention_82463372083671
// MI455X (gfx1250) — matrix operand order repaired and checked by reading, not compiled, not run on hardware
//
#include <hip/hip_runtime.h>

typedef __attribute__((ext_vector_type(16))) __bf16 v16bf;
typedef __attribute__((ext_vector_type(8)))  float  v8f;

union BF16x16 {
    v16bf v;
    uint4 q[2];
};

__device__ __forceinline__ unsigned short f2bf(float f) {
    unsigned u = __builtin_bit_cast(unsigned, f);
    unsigned r = u + 0x7fffu + ((u >> 16) & 1u);   // round-to-nearest-even
    return (unsigned short)(r >> 16);
}
__device__ __forceinline__ unsigned pack2(float a, float b) {
    return (unsigned)f2bf(a) | ((unsigned)f2bf(b) << 16);
}

#define LSTR 40   // LDS row stride in bf16 elements (80B, 16B aligned, depads banks)

// ---------------------------------------------------------------------------
// Kernel 1: qkv = x @ W_attn + b_attn   (M=4096, N=3072, K=1024)
// writes Q,K as bf16 [B,H,S,Dh] and V as bf16 [B,H,Dh,S] (transposed).
// Register-staged pipeline: next K-slice global loads overlap current WMMAs.
// ---------------------------------------------------------------------------
__global__ __launch_bounds__(256) void qkv_gemm_kernel(
    const float* __restrict__ X, const float* __restrict__ W,
    const float* __restrict__ bias,
    unsigned short* __restrict__ qb, unsigned short* __restrict__ kb,
    unsigned short* __restrict__ vt)
{
    const int K = 1024, N = 3072;
    __shared__ alignas(16) unsigned short ldsA[128 * LSTR];  // [m][k]
    __shared__ alignas(16) unsigned short ldsB[128 * LSTR];  // [n][k] (transposed)

    const int tid = threadIdx.x;
    const int l  = tid & 31, w = tid >> 5;
    const int ln = l & 15, lh = (l >> 4) & 1;
    const int wm = w >> 2, wn = w & 3;               // 2x4 wave grid
    const int row0 = blockIdx.y * 128;
    const int col0 = blockIdx.x * 128;

    v8f acc[8];
    #pragma unroll
    for (int i = 0; i < 8; ++i)
        #pragma unroll
        for (int r = 0; r < 8; ++r) acc[i][r] = 0.0f;

    const int ar = tid >> 1, ac = (tid & 1) * 16;    // A tile: 128x32
    const int bk = tid >> 3, bc = (tid & 7) * 16;    // B tile: 32x128

    const float* aptr = X + (size_t)(row0 + ar) * K + ac;
    const float* bptr = W + (size_t)bk * N + col0 + bc;

    float4 fa[4], fb[4];
    #pragma unroll
    for (int c = 0; c < 4; ++c) fa[c] = ((const float4*)aptr)[c];
    #pragma unroll
    for (int c = 0; c < 4; ++c) fb[c] = ((const float4*)bptr)[c];

    for (int k0 = 0; k0 < K; k0 += 32) {
        {   // A: regs (f32) -> bf16 LDS, row major
            uint4* dst = (uint4*)&ldsA[ar * LSTR + ac];
            dst[0] = make_uint4(pack2(fa[0].x,fa[0].y), pack2(fa[0].z,fa[0].w),
                                pack2(fa[1].x,fa[1].y), pack2(fa[1].z,fa[1].w));
            dst[1] = make_uint4(pack2(fa[2].x,fa[2].y), pack2(fa[2].z,fa[2].w),
                                pack2(fa[3].x,fa[3].y), pack2(fa[3].z,fa[3].w));
        }
        {   // B: regs (f32) -> bf16 LDS, transposed to [n][k]
            float vv[16] = {fb[0].x,fb[0].y,fb[0].z,fb[0].w, fb[1].x,fb[1].y,fb[1].z,fb[1].w,
                            fb[2].x,fb[2].y,fb[2].z,fb[2].w, fb[3].x,fb[3].y,fb[3].z,fb[3].w};
            #pragma unroll
            for (int j = 0; j < 16; ++j)
                ldsB[(bc + j) * LSTR + bk] = f2bf(vv[j]);
        }
        __syncthreads();

        // prefetch next K-slice (clamped on last iter: harmless re-read)
        const int kn = (k0 + 32 < K) ? (k0 + 32) : k0;
        #pragma unroll
        for (int c = 0; c < 4; ++c) fa[c] = ((const float4*)(aptr + kn))[c];
        #pragma unroll
        for (int c = 0; c < 4; ++c) fb[c] = ((const float4*)(bptr + (size_t)kn * N))[c];

        BF16x16 a[4], b[2];
        #pragma unroll
        for (int i = 0; i < 4; ++i) {   // A frag: lanes hold K 0..15 (lo) or 16..31 (hi)
            const unsigned short* p = &ldsA[(wm*64 + i*16 + ln) * LSTR + lh*16];
            a[i].q[0] = ((const uint4*)p)[0];
            a[i].q[1] = *(const uint4*)(p + 8);
        }
        #pragma unroll
        for (int j = 0; j < 2; ++j) {   // B frag: lanes hold K 0..15 (lo) or 16..31 (hi)
            const unsigned short* p = &ldsB[(wn*32 + j*16 + ln) * LSTR + lh*16];
            b[j].q[0] = ((const uint4*)p)[0];
            b[j].q[1] = *(const uint4*)(p + 8);
        }
        #pragma unroll
        for (int i = 0; i < 4; ++i)
            #pragma unroll
            for (int j = 0; j < 2; ++j)
                acc[i*2+j] = __builtin_amdgcn_wmma_f32_16x16x32_bf16(
                    false, a[i].v, false, b[j].v, (short)0, acc[i*2+j], false, false);
        __syncthreads();
    }

    // epilogue: + bias, scatter into Q / K / V^T (bf16)
    #pragma unroll
    for (int j = 0; j < 2; ++j) {
        const int col  = col0 + wn*32 + j*16 + ln;
        const float bv = bias[col];
        const int part = col >> 10;            // 0=q 1=k 2=v (uniform per 16-wide tile)
        const int dcol = col & 1023;
        const int h = dcol >> 6, dh = dcol & 63;
        #pragma unroll
        for (int i = 0; i < 4; ++i) {
            const int rb = row0 + wm*64 + i*16 + lh*8;
            #pragma unroll
            for (int r = 0; r < 8; ++r) {
                const int grow = rb + r;
                const int bb = grow >> 11, s = grow & 2047;
                const unsigned short val = f2bf(acc[i*2+j][r] + bv);
                if (part == 0)
                    qb[(((size_t)(bb*16 + h)) * 2048 + s) * 64 + dh] = val;
                else if (part == 1)
                    kb[(((size_t)(bb*16 + h)) * 2048 + s) * 64 + dh] = val;
                else
                    vt[(((size_t)(bb*16 + h)) * 64 + dh) * 2048 + s] = val;
            }
        }
    }
}

// ---------------------------------------------------------------------------
// Kernel 2: causal flash attention, one 16-row q-tile per wave.
// K/V fragments stream from global (L2-resident: whole K/V = 16MB << 192MB L2).
// ---------------------------------------------------------------------------
__global__ __launch_bounds__(256) void attn_kernel(
    const unsigned short* __restrict__ qb, const unsigned short* __restrict__ kb,
    const unsigned short* __restrict__ vt, unsigned short* __restrict__ ob)
{
    __shared__ alignas(16) unsigned short lds_p[8][16 * LSTR];  // per-wave P scratch

    const int bh   = blockIdx.x;           // b*16+h, 0..31
    const int qblk = blockIdx.y;           // 0..15
    const int l  = threadIdx.x & 31, w = threadIdx.x >> 5;
    const int ln = l & 15, lh = (l >> 4) & 1;
    const int q0 = (qblk * 8 + w) * 16;

    const unsigned short* Q  = qb + (size_t)bh * 2048 * 64;
    const unsigned short* Kp = kb + (size_t)bh * 2048 * 64;
    const unsigned short* Vt = vt + (size_t)bh * 64 * 2048;
    unsigned short* Pl = lds_p[w];

    // Q A-fragments for d-chunks {0..31},{32..63}
    BF16x16 aq[2];
    {
        const unsigned short* qrow = Q + (size_t)(q0 + ln) * 64;
        #pragma unroll
        for (int i = 0; i < 2; ++i) {
            const unsigned short* p = qrow + i*32 + lh*16;
            aq[i].q[0] = ((const uint4*)p)[0];
            aq[i].q[1] = *(const uint4*)(p + 8);
        }
    }

    v8f acc[4];
    #pragma unroll
    for (int t = 0; t < 4; ++t)
        #pragma unroll
        for (int r = 0; r < 8; ++r) acc[t][r] = 0.0f;
    float mrow[8], lrow[8];
    #pragma unroll
    for (int r = 0; r < 8; ++r) { mrow[r] = -__builtin_inff(); lrow[r] = 0.0f; }

    const float scale = 0.125f;                 // 1/sqrt(64)
    const int kend = (q0 + 15) >> 5;            // inclusive 32-wide k-tile index

    for (int kt = 0; kt <= kend; ++kt) {
        const int k0 = kt * 32;
        // scores: two 16x16 tiles, K-dim=64 -> 2 WMMAs each
        v8f s[2];
        #pragma unroll
        for (int j = 0; j < 2; ++j) {
            BF16x16 bk0, bk1;
            const unsigned short* krow = Kp + (size_t)(k0 + j*16 + ln) * 64 + lh*16;
            bk0.q[0] = ((const uint4*)krow)[0];
            bk0.q[1] = *(const uint4*)(krow + 8);
            bk1.q[0] = *(const uint4*)(krow + 32);
            bk1.q[1] = *(const uint4*)(krow + 40);
            v8f z;
            #pragma unroll
            for (int r = 0; r < 8; ++r) z[r] = 0.0f;
            z = __builtin_amdgcn_wmma_f32_16x16x32_bf16(false, aq[0].v, false, bk0.v, (short)0, z, false, false);
            z = __builtin_amdgcn_wmma_f32_16x16x32_bf16(false, aq[1].v, false, bk1.v, (short)0, z, false, false);
            s[j] = z;
        }
        // causal mask + online softmax (row state replicated across each half-wave)
        #pragma unroll
        for (int r = 0; r < 8; ++r) {
            const int grow = q0 + r + lh*8;
            float t0 = s[0][r] * scale;
            float t1 = s[1][r] * scale;
            if (k0 + ln      > grow) t0 = -__builtin_inff();
            if (k0 + 16 + ln > grow) t1 = -__builtin_inff();
            float mx = fmaxf(t0, t1);
            #pragma unroll
            for (int d = 8; d >= 1; d >>= 1) mx = fmaxf(mx, __shfl_xor(mx, d, 32));
            const float mnew = fmaxf(mrow[r], mx);
            const float al = __expf(mrow[r] - mnew);
            const float p0 = __expf(t0 - mnew);
            const float p1 = __expf(t1 - mnew);
            float rs = p0 + p1;
            #pragma unroll
            for (int d = 8; d >= 1; d >>= 1) rs += __shfl_xor(rs, d, 32);
            lrow[r] = lrow[r] * al + rs;
            mrow[r] = mnew;
            #pragma unroll
            for (int t = 0; t < 4; ++t) acc[t][r] *= al;
            Pl[(r + lh*8) * LSTR + ln]      = f2bf(p0);
            Pl[(r + lh*8) * LSTR + 16 + ln] = f2bf(p1);
        }
        __builtin_amdgcn_wave_barrier();   // LDS is in-order per wave; pin scheduling
        BF16x16 ap;                        // P as 16x32 A-fragment
        {
            const unsigned short* p = &Pl[ln * LSTR + lh*16];
            ap.q[0] = ((const uint4*)p)[0];
            ap.q[1] = *(const uint4*)(p + 8);
        }
        __builtin_amdgcn_wave_barrier();
        // O += P @ V  (V^T rows are contiguous B-fragments)
        #pragma unroll
        for (int t = 0; t < 4; ++t) {
            BF16x16 bv;
            const unsigned short* vrow = Vt + (size_t)(t*16 + ln) * 2048 + k0 + lh*16;
            bv.q[0] = ((const uint4*)vrow)[0];
            bv.q[1] = *(const uint4*)(vrow + 8);
            acc[t] = __builtin_amdgcn_wmma_f32_16x16x32_bf16(
                false, ap.v, false, bv.v, (short)0, acc[t], false, false);
        }
    }

    // finalize: /= rowsum, store bf16 to [B,S,D] buffer for the output GEMM
    const int bb = bh >> 4, h = bh & 15;
    #pragma unroll
    for (int r = 0; r < 8; ++r) {
        const float inv = 1.0f / lrow[r];
        const int grow = q0 + r + lh*8;
        unsigned short* orow = ob + ((size_t)(bb*2048 + grow)) * 1024 + h*64;
        #pragma unroll
        for (int t = 0; t < 4; ++t)
            orow[t*16 + ln] = f2bf(acc[t][r] * inv);
    }
}

// ---------------------------------------------------------------------------
// Kernel 3: out = attn_out @ W_proj + b_proj   (M=4096, N=1024, K=1024, f32 out)
// Same register-staged pipeline as kernel 1; A is already bf16.
// ---------------------------------------------------------------------------
__global__ __launch_bounds__(256) void proj_gemm_kernel(
    const unsigned short* __restrict__ A, const float* __restrict__ W,
    const float* __restrict__ bias, float* __restrict__ out)
{
    const int K = 1024, N = 1024;
    __shared__ alignas(16) unsigned short ldsA[128 * LSTR];
    __shared__ alignas(16) unsigned short ldsB[128 * LSTR];

    const int tid = threadIdx.x;
    const int l  = tid & 31, w = tid >> 5;
    const int ln = l & 15, lh = (l >> 4) & 1;
    const int wm = w >> 2, wn = w & 3;
    const int row0 = blockIdx.y * 128;
    const int col0 = blockIdx.x * 128;

    v8f acc[8];
    #pragma unroll
    for (int i = 0; i < 8; ++i)
        #pragma unroll
        for (int r = 0; r < 8; ++r) acc[i][r] = 0.0f;

    const int ar = tid >> 1, ac = (tid & 1) * 16;
    const int bk = tid >> 3, bc = (tid & 7) * 16;

    const unsigned short* aptr = A + (size_t)(row0 + ar) * K + ac;
    const float*          bptr = W + (size_t)bk * N + col0 + bc;

    uint4  ua[2];
    float4 fb[4];
    #pragma unroll
    for (int c = 0; c < 2; ++c) ua[c] = ((const uint4*)aptr)[c];
    #pragma unroll
    for (int c = 0; c < 4; ++c) fb[c] = ((const float4*)bptr)[c];

    for (int k0 = 0; k0 < K; k0 += 32) {
        {   // A already bf16: straight 32B store into LDS
            uint4* dst = (uint4*)&ldsA[ar * LSTR + ac];
            dst[0] = ua[0];
            dst[1] = ua[1];
        }
        {   // W_proj: regs (f32) -> bf16, transposed
            float vv[16] = {fb[0].x,fb[0].y,fb[0].z,fb[0].w, fb[1].x,fb[1].y,fb[1].z,fb[1].w,
                            fb[2].x,fb[2].y,fb[2].z,fb[2].w, fb[3].x,fb[3].y,fb[3].z,fb[3].w};
            #pragma unroll
            for (int j = 0; j < 16; ++j)
                ldsB[(bc + j) * LSTR + bk] = f2bf(vv[j]);
        }
        __syncthreads();

        // prefetch next K-slice (clamped on last iter)
        const int kn = (k0 + 32 < K) ? (k0 + 32) : k0;
        #pragma unroll
        for (int c = 0; c < 2; ++c) ua[c] = ((const uint4*)(aptr + kn))[c];
        #pragma unroll
        for (int c = 0; c < 4; ++c) fb[c] = ((const float4*)(bptr + (size_t)kn * N))[c];

        BF16x16 a[4], b[2];
        #pragma unroll
        for (int i = 0; i < 4; ++i) {
            const unsigned short* p = &ldsA[(wm*64 + i*16 + ln) * LSTR + lh*16];
            a[i].q[0] = ((const uint4*)p)[0];
            a[i].q[1] = *(const uint4*)(p + 8);
        }
        #pragma unroll
        for (int j = 0; j < 2; ++j) {
            const unsigned short* p = &ldsB[(wn*32 + j*16 + ln) * LSTR + lh*16];
            b[j].q[0] = ((const uint4*)p)[0];
            b[j].q[1] = *(const uint4*)(p + 8);
        }
        #pragma unroll
        for (int i = 0; i < 4; ++i)
            #pragma unroll
            for (int j = 0; j < 2; ++j)
                acc[i*2+j] = __builtin_amdgcn_wmma_f32_16x16x32_bf16(
                    false, a[i].v, false, b[j].v, (short)0, acc[i*2+j], false, false);
        __syncthreads();
    }

    #pragma unroll
    for (int j = 0; j < 2; ++j) {
        const int col = col0 + wn*32 + j*16 + ln;
        const float bv = bias[col];
        #pragma unroll
        for (int i = 0; i < 4; ++i) {
            const int rb = row0 + wm*64 + i*16 + lh*8;
            #pragma unroll
            for (int r = 0; r < 8; ++r)
                out[(size_t)(rb + r) * N + col] = acc[i*2+j][r] + bv;
        }
    }
}

// ---------------------------------------------------------------------------
extern "C" void kernel_launch(void* const* d_in, const int* in_sizes, int n_in,
                              void* d_out, int out_size, void* d_ws, size_t ws_size,
                              hipStream_t stream) {
    const float* x      = (const float*)d_in[0];   // [2,2048,1024]
    const float* W_attn = (const float*)d_in[1];   // [1024,3072]
    const float* b_attn = (const float*)d_in[2];   // [3072]
    const float* W_proj = (const float*)d_in[3];   // [1024,1024]
    const float* b_proj = (const float*)d_in[4];   // [1024]
    float* out = (float*)d_out;                    // [2,2048,1024] f32

    const size_t qkv_elems = (size_t)2 * 16 * 2048 * 64;   // 4,194,304
    unsigned short* qb = (unsigned short*)d_ws;            // Q  [B,H,S,Dh] bf16
    unsigned short* kb = qb + qkv_elems;                   // K  [B,H,S,Dh] bf16
    unsigned short* vt = kb + qkv_elems;                   // V^T[B,H,Dh,S] bf16
    unsigned short* ob = vt + qkv_elems;                   // O  [B,S,D]    bf16

    dim3 g1(3072 / 128, 4096 / 128);   // (24, 32)
    qkv_gemm_kernel<<<g1, 256, 0, stream>>>(x, W_attn, b_attn, qb, kb, vt);

    dim3 g2(32, 16);                   // (B*H, S / (16*8))
    attn_kernel<<<g2, 256, 0, stream>>>(qb, kb, vt, ob);

    dim3 g3(1024 / 128, 4096 / 128);   // (8, 32)
    proj_gemm_kernel<<<g3, 256, 0, stream>>>(ob, W_proj, b_proj, out);
}
